// GraphEncoder_23081154249451
// MI455X (gfx1250) — compile-verified, matrix operand order checked by reading, not run on hardware
//
#include <hip/hip_runtime.h>
#include <hip/hip_bf16.h>

typedef _Float16 v16h __attribute__((ext_vector_type(16)));
typedef _Float16 v8h  __attribute__((ext_vector_type(8)));
typedef float    v8f  __attribute__((ext_vector_type(8)));

#define NEG_SLOPE 0.2f

// ---------------- utility ----------------
__global__ void fill_kernel(float* __restrict__ p, float v, size_t n) {
  for (size_t i = (size_t)blockIdx.x * blockDim.x + threadIdx.x; i < n;
       i += (size_t)gridDim.x * blockDim.x)
    p[i] = v;
}

__device__ __forceinline__ void atomicMaxF(float* addr, float val) {
  unsigned int* ai = reinterpret_cast<unsigned int*>(addr);
  unsigned int cur = __float_as_uint(*addr);
  while (__uint_as_float(cur) < val) {
    unsigned int old = atomicCAS(ai, cur, __float_as_uint(val));
    if (old == cur) break;
    cur = old;
  }
}

// ---------------- layer 1: h1 = x @ W1 ; al_s/al_d dots ----------------
// block = 256 threads (one per output feature), grid = N nodes.
__global__ void gat1_node(const float* __restrict__ x, const float* __restrict__ W,
                          const float* __restrict__ a_s, const float* __restrict__ a_d,
                          float* __restrict__ h, float* __restrict__ als,
                          float* __restrict__ ald, int N) {
  __shared__ float s0[256];
  __shared__ float s1[256];
  const int n = blockIdx.x;
  const int f = threadIdx.x;
  float acc = 0.f;
#pragma unroll
  for (int k = 0; k < 5; ++k) acc += x[n * 5 + k] * W[k * 256 + f];
  h[(size_t)n * 256 + f] = acc;
  s0[f] = acc * a_s[f];
  s1[f] = acc * a_d[f];
  __syncthreads();
  const int head = f >> 6, c = f & 63;
  for (int off = 32; off > 0; off >>= 1) {
    if (c < off) { s0[f] += s0[f + off]; s1[f] += s1[f + off]; }
    __syncthreads();
  }
  if (c == 0) { als[(size_t)n * 4 + head] = s0[f]; ald[(size_t)n * 4 + head] = s1[f]; }
}

// ---------------- layer 2 attention dots (C = 64, H = 1) ----------------
__global__ void attn_dots64(const float* __restrict__ h, const float* __restrict__ a_s,
                            const float* __restrict__ a_d, float* __restrict__ als,
                            float* __restrict__ ald, int N) {
  __shared__ float s0[64];
  __shared__ float s1[64];
  const int n = blockIdx.x, c = threadIdx.x;
  const float v = h[(size_t)n * 64 + c];
  s0[c] = v * a_s[c];
  s1[c] = v * a_d[c];
  __syncthreads();
  for (int off = 32; off > 0; off >>= 1) {
    if (c < off) { s0[c] += s0[c + off]; s1[c] += s1[c + off]; }
    __syncthreads();
  }
  if (c == 0) { als[n] = s0[0]; ald[n] = s1[0]; }
}

// ---------------- edge logits + segment max ----------------
__global__ void edge_logits(const int* __restrict__ ei, const float* __restrict__ als,
                            const float* __restrict__ ald, float* __restrict__ e,
                            float* __restrict__ m, int E, int N, int H) {
  const long eid = (long)blockIdx.x * blockDim.x + threadIdx.x;
  if (eid >= (long)E + N) return;
  int s, d;
  if (eid < E) { s = ei[eid]; d = ei[(size_t)E + eid]; } else { s = d = (int)(eid - E); }
  for (int hh = 0; hh < H; ++hh) {
    float v = als[(size_t)s * H + hh] + ald[(size_t)d * H + hh];
    v = v > 0.f ? v : NEG_SLOPE * v;
    e[(size_t)eid * H + hh] = v;
    atomicMaxF(&m[(size_t)d * H + hh], v);
  }
}

// ---------------- edge exp + segment sum ----------------
__global__ void edge_exp(const int* __restrict__ ei, float* __restrict__ e,
                         const float* __restrict__ m, float* __restrict__ denom,
                         int E, int N, int H) {
  const long eid = (long)blockIdx.x * blockDim.x + threadIdx.x;
  if (eid >= (long)E + N) return;
  const int d = (eid < E) ? ei[(size_t)E + eid] : (int)(eid - E);
  for (int hh = 0; hh < H; ++hh) {
    const float ex = __expf(e[(size_t)eid * H + hh] - m[(size_t)d * H + hh]);
    e[(size_t)eid * H + hh] = ex;
    atomicAdd(&denom[(size_t)d * H + hh], ex);
  }
}

// ---------------- edge aggregation: out[dst] += alpha * h[src] ----------------
// one wave (32 lanes) per edge; F channels strided by 32 across lanes.
__global__ void aggregate(const float* __restrict__ h, const float* __restrict__ e,
                          const float* __restrict__ denom, const int* __restrict__ ei,
                          float* __restrict__ out, int E, int N, int H, int F) {
  const int lane = threadIdx.x & 31;
  const long eid = (long)blockIdx.x * (blockDim.x >> 5) + (threadIdx.x >> 5);
  if (eid >= (long)E + N) return;
  int s, d;
  if (eid < E) { s = ei[eid]; d = ei[(size_t)E + eid]; } else { s = d = (int)(eid - E); }
  float alpha[4];
  for (int hh = 0; hh < H; ++hh)
    alpha[hh] = e[(size_t)eid * H + hh] / denom[(size_t)d * H + hh];
  const int CH = F / H;
  for (int c = lane; c < F; c += 32) {
    const float v = alpha[c / CH] * h[(size_t)s * F + c];
    atomicAdd(&out[(size_t)d * F + c], v);
  }
}

// ---------------- bias + relu + f32->f16 convert (out-of-place) ----------------
__global__ void relu_bias_cvt(const float* __restrict__ p, const float* __restrict__ b,
                              _Float16* __restrict__ q, size_t total, int F) {
  for (size_t i = (size_t)blockIdx.x * blockDim.x + threadIdx.x; i < total;
       i += (size_t)gridDim.x * blockDim.x) {
    const float v = p[i] + b[i & (size_t)(F - 1)];
    q[i] = (_Float16)(v > 0.f ? v : 0.f);
  }
}

// ---------------- W2 [256,64] f32 -> W2t [64,256] f16 ----------------
__global__ void transpose_w2(const float* __restrict__ W, _Float16* __restrict__ Wt) {
  const int i = blockIdx.x * blockDim.x + threadIdx.x;  // i over 64*256
  if (i >= 64 * 256) return;
  const int n = i >> 8, k = i & 255;
  Wt[i] = (_Float16)W[k * 64 + n];
}

// ---------------- layer 2 GEMM: [M,256](f16) x [256,64](f16,B pre-transposed) ----
// block = 128 threads = 4 waves (one wave per 16-wide N tile), grid = full M tiles.
__global__ void gemm2_wmma(const _Float16* __restrict__ A, const _Float16* __restrict__ Bt,
                           float* __restrict__ D) {
  const int K = 256, NC = 64;
  const int wave = threadIdx.x >> 5;
  const int lane = threadIdx.x & 31;
  const int m0 = blockIdx.x * 16;
  const int n0 = wave * 16;
  const int r = lane & 15;
  const bool hi = lane >= 16;
  const int row = m0 + r;
  const int col = n0 + r;
  const _Float16* arow = A + (size_t)row * K;
  const _Float16* brow = Bt + (size_t)col * K;
  v8f acc = {};
#pragma unroll
  for (int k0 = 0; k0 < K; k0 += 32) {
    // 16-bit A VGPR layout: lane<16 holds K {k0..k0+7, k0+16..k0+23} of row r,
    // lane>=16 holds K {k0+8..k0+15, k0+24..k0+31}.  B mirrors with col striped.
    const int ka = k0 + (hi ? 8 : 0);
    const v8h a_lo = *(const v8h*)(arow + ka);
    const v8h a_hi = *(const v8h*)(arow + ka + 16);
    const v8h b_lo = *(const v8h*)(brow + ka);
    const v8h b_hi = *(const v8h*)(brow + ka + 16);
    const v16h av = __builtin_shufflevector(a_lo, a_hi, 0, 1, 2, 3, 4, 5, 6, 7,
                                            8, 9, 10, 11, 12, 13, 14, 15);
    const v16h bv = __builtin_shufflevector(b_lo, b_hi, 0, 1, 2, 3, 4, 5, 6, 7,
                                            8, 9, 10, 11, 12, 13, 14, 15);
    acc = __builtin_amdgcn_wmma_f32_16x16x32_f16(false, av, false, bv, (short)0, acc,
                                                 false, false);
  }
  // D layout: VGPR rr -> row m0+rr (lanes 0-15) / m0+8+rr (lanes 16-31), col n0+r
  float* dbase = D + (size_t)(m0 + (hi ? 8 : 0)) * NC + col;
#pragma unroll
  for (int rr = 0; rr < 8; ++rr) dbase[(size_t)rr * NC] = acc[rr];
}

// scalar fallback for M % 16 remainder rows (zero blocks when M is a multiple of 16)
__global__ void gemm2_rem(const _Float16* __restrict__ A, const _Float16* __restrict__ Bt,
                          float* __restrict__ D, int mstart) {
  const int m = mstart + blockIdx.x;
  const int c = threadIdx.x;  // 0..63
  float acc = 0.f;
  for (int k = 0; k < 256; ++k)
    acc += (float)A[(size_t)m * 256 + k] * (float)Bt[(size_t)c * 256 + k];
  D[(size_t)m * 64 + c] = acc;
}

// ---------------- final column-sum over nodes ----------------
__global__ void colsum(const float* __restrict__ a, float* __restrict__ out, int N, int C) {
  const int tid = blockIdx.x * blockDim.x + threadIdx.x;
  const int c = tid & (C - 1);
  const int row0 = tid / C;
  const int rstep = (gridDim.x * blockDim.x) / C;
  float s = 0.f;
  for (int r2 = row0; r2 < N; r2 += rstep) s += a[(size_t)r2 * C + c];
  atomicAdd(&out[c], s);
}

__global__ void finalize_kernel(float* __restrict__ out, const float* __restrict__ b,
                                float invN, int C) {
  const int c = threadIdx.x;
  if (c < C) out[c] = out[c] * invN + b[c];
}

// ---------------- host ----------------
extern "C" void kernel_launch(void* const* d_in, const int* in_sizes, int n_in,
                              void* d_out, int out_size, void* d_ws, size_t ws_size,
                              hipStream_t stream) {
  const float* x   = (const float*)d_in[0];
  const int*   ei  = (const int*)d_in[1];
  const float* W1  = (const float*)d_in[2];
  const float* as1 = (const float*)d_in[3];
  const float* ad1 = (const float*)d_in[4];
  const float* b1  = (const float*)d_in[5];
  const float* W2  = (const float*)d_in[6];
  const float* as2 = (const float*)d_in[7];
  const float* ad2 = (const float*)d_in[8];
  const float* b2  = (const float*)d_in[9];
  float* out = (float*)d_out;

  const int N = in_sizes[0] / 5;
  const int E = in_sizes[1] / 2;
  const long Etot = (long)E + N;

  // workspace layout (float units), with lifetime-checked reuse:
  //  region A [N*256 floats]: h1 (layer1) -> { x2h (N*256 f16 = N*128 f32) ; h2 (N*64 f32) }
  //  region B [N*256 floats]: out1 (layer1, read by relu_cvt) -> out2 (N*64 f32)
  float* ws    = (float*)d_ws;
  float* h1    = ws;
  float* o1    = h1 + (size_t)N * 256;
  float* att   = o1 + (size_t)N * 256;        // 16*N scalars
  float* als_1 = att;
  float* ald_1 = att + 4 * (size_t)N;
  float* m1    = att + 8 * (size_t)N;
  float* dn1   = att + 12 * (size_t)N;
  float* e1    = att + 16 * (size_t)N;        // Etot*4 floats; later e2 (Etot)
  float* w2t_f = e1 + 4 * (size_t)Etot;       // 64*256 halves = 8192 floats

  _Float16* x2h = (_Float16*)h1;              // N*256 halves (h1 dead by then)
  float*    h2  = h1 + (size_t)N * 128;       // N*64 floats, inside region A
  _Float16* w2t = (_Float16*)w2t_f;
  // layer-2 scalars reuse layer-1 attn region (dead by then)
  float* als_2 = att;
  float* ald_2 = att + (size_t)N;
  float* m2    = att + 2 * (size_t)N;
  float* dn2   = att + 3 * (size_t)N;
  float* o2    = o1;
  float* e2    = e1;

  const float NEG_INF = -__builtin_inff();
  auto gsz = [](size_t n, int b) { return (unsigned)((n + b - 1) / b); };

  // ---- layer 1 ----
  fill_kernel<<<gsz(4 * (size_t)N, 256), 256, 0, stream>>>(m1, NEG_INF, 4 * (size_t)N);
  fill_kernel<<<gsz(4 * (size_t)N, 256), 256, 0, stream>>>(dn1, 0.f, 4 * (size_t)N);
  fill_kernel<<<2048, 256, 0, stream>>>(o1, 0.f, (size_t)N * 256);
  gat1_node<<<N, 256, 0, stream>>>(x, W1, as1, ad1, h1, als_1, ald_1, N);
  edge_logits<<<gsz(Etot, 256), 256, 0, stream>>>(ei, als_1, ald_1, e1, m1, E, N, 4);
  edge_exp<<<gsz(Etot, 256), 256, 0, stream>>>(ei, e1, m1, dn1, E, N, 4);
  aggregate<<<gsz(Etot, 8), 256, 0, stream>>>(h1, e1, dn1, ei, o1, E, N, 4, 256);
  // x2h = relu(out1 + b1) as f16 (h1 region; h1 is dead here)
  relu_bias_cvt<<<2048, 256, 0, stream>>>(o1, b1, x2h, (size_t)N * 256, 256);

  // ---- layer 2 ----
  transpose_w2<<<gsz(64 * 256, 256), 256, 0, stream>>>(W2, w2t);
  const int Mfull = N / 16, Mrem = N % 16;
  gemm2_wmma<<<Mfull, 128, 0, stream>>>(x2h, w2t, h2);
  if (Mrem) gemm2_rem<<<Mrem, 64, 0, stream>>>(x2h, w2t, h2, Mfull * 16);
  attn_dots64<<<N, 64, 0, stream>>>(h2, as2, ad2, als_2, ald_2, N);
  fill_kernel<<<gsz((size_t)N, 256), 256, 0, stream>>>(m2, NEG_INF, (size_t)N);
  fill_kernel<<<gsz((size_t)N, 256), 256, 0, stream>>>(dn2, 0.f, (size_t)N);
  fill_kernel<<<gsz((size_t)N * 64, 256), 256, 0, stream>>>(o2, 0.f, (size_t)N * 64);
  edge_logits<<<gsz(Etot, 256), 256, 0, stream>>>(ei, als_2, ald_2, e2, m2, E, N, 1);
  edge_exp<<<gsz(Etot, 256), 256, 0, stream>>>(ei, e2, m2, dn2, E, N, 1);
  aggregate<<<gsz(Etot, 8), 256, 0, stream>>>(h2, e2, dn2, ei, o2, E, N, 1, 64);

  // ---- final mean over nodes + bias ----
  fill_kernel<<<1, 64, 0, stream>>>(out, 0.f, 64);
  colsum<<<128, 256, 0, stream>>>(o2, out, N, 64);
  finalize_kernel<<<1, 64, 0, stream>>>(out, b2, 1.0f / (float)N, 64);
}
